// LagrangianLayer_20203526160987
// MI455X (gfx1250) — matrix operand order repaired and checked by reading, not compiled, not run on hardware
//
#include <hip/hip_runtime.h>

// Problem constants (match reference)
#define BB 2048
#define DD 512
#define NN 512
#define QQ 32

typedef __bf16 bf16t;
typedef bf16t v16bf __attribute__((ext_vector_type(16)));
typedef bf16t v8bf  __attribute__((ext_vector_type(8)));
typedef float v8f   __attribute__((ext_vector_type(8)));

// f32 -> bf16, round-to-nearest-even (inputs are finite normals)
__device__ __forceinline__ unsigned short f2bf(float f) {
    unsigned int u = __float_as_uint(f);
    u += 0x7FFFu + ((u >> 16) & 1u);
    return (unsigned short)(u >> 16);
}

// ---------------------------------------------------------------------------
// Kernel 0a: swizzle weight matrix K[D][N] into
//   Asw: WMMA A-operand layout of K^T   (A[m=n][k=d])  -- for backward
//   Bsw: WMMA B-operand layout of K     (B[k=d][n])    -- for forward
// Tile grid: 32 n-tiles x 16 k-chunks; per tile 32 lanes x 16 bf16 (32B/lane).
// ---------------------------------------------------------------------------
__global__ void __launch_bounds__(256)
swizzle_weights(const float* __restrict__ K,
                unsigned short* __restrict__ Asw,
                unsigned short* __restrict__ Bsw) {
    int t    = blockIdx.x * 256 + threadIdx.x;   // 0 .. 262143
    int e    = t & 15;
    int lane = (t >> 4) & 31;
    int kc   = (t >> 9) & 15;
    int nt   = t >> 13;                          // 0 .. 31
    int m    = lane & 15;
    int half = lane >> 4;

    // A-operand (16x32, 16-bit), contiguous k order used here: lanes 0-15 K=0..15, lanes 16-31 K=16..31
    int kA = half * 16 + e;
    int dA = kc * 32 + kA;
    int nA = nt * 16 + m;
    Asw[t] = f2bf(K[dA * NN + nA]);

    // B-operand (32x16, 16-bit), same contiguous k order as A: lanes 0-15 K=0..15, lanes 16-31 K=16..31
    int dB = kc * 32 + half * 16 + e;
    int nB = nt * 16 + m;
    Bsw[t] = f2bf(K[dB * NN + nB]);
}

// Kernel 0b: swizzle inputs[B][D] into WMMA A-operand layout (A[m=b][k=d]).
__global__ void __launch_bounds__(256)
swizzle_inputs(const float* __restrict__ X, unsigned short* __restrict__ Isw) {
    int t    = blockIdx.x * 256 + threadIdx.x;   // 0 .. 1048575
    int e    = t & 15;
    int lane = (t >> 4) & 31;
    int kc   = (t >> 9) & 15;
    int bt   = t >> 13;                          // 0 .. 127
    int m    = lane & 15;
    int half = lane >> 4;
    int k    = half * 16 + e;
    int d    = kc * 32 + k;
    int b    = bt * 16 + m;
    Isw[t] = f2bf(X[b * DD + d]);
}

// ---------------------------------------------------------------------------
// Forward: output = relu(X @ K + bias), bf16 WMMA with f32 accumulate.
// Grid: (B/64, N/64) blocks of 256 threads. Wave handles 1 n-tile x 2 b-tiles.
// ---------------------------------------------------------------------------
__global__ void __launch_bounds__(256)
forward_gemm(const unsigned short* __restrict__ Isw,
             const unsigned short* __restrict__ Bsw,
             const float* __restrict__ bias,
             float* __restrict__ output) {
    int tid  = threadIdx.x;
    int w    = tid >> 5;
    int lane = tid & 31;
    int half = lane >> 4;
    int nl   = lane & 15;

    int nTile  = blockIdx.y * 4 + (w & 3);        // 0..31 (global n tile)
    int bTile0 = blockIdx.x * 4 + (w >> 2) * 2;   // global b tile (16 rows)

    v8f acc0 = {0.f, 0.f, 0.f, 0.f, 0.f, 0.f, 0.f, 0.f};
    v8f acc1 = {0.f, 0.f, 0.f, 0.f, 0.f, 0.f, 0.f, 0.f};

    for (int kc = 0; kc < 16; ++kc) {
        v16bf bfrag = *(const v16bf*)(Bsw + (((nTile * 16 + kc) * 32 + lane) << 4));
        v16bf a0    = *(const v16bf*)(Isw + (((bTile0 * 16 + kc) * 32 + lane) << 4));
        v16bf a1    = *(const v16bf*)(Isw + ((((bTile0 + 1) * 16 + kc) * 32 + lane) << 4));
        acc0 = __builtin_amdgcn_wmma_f32_16x16x32_bf16(false, a0, false, bfrag,
                                                       (short)0, acc0, false, false);
        acc1 = __builtin_amdgcn_wmma_f32_16x16x32_bf16(false, a1, false, bfrag,
                                                       (short)0, acc1, false, false);
    }

    int n = nTile * 16 + nl;
    float bv = bias[n];
#pragma unroll
    for (int r = 0; r < 8; ++r) {
        int b0 = bTile0 * 16 + r + 8 * half;
        float v0 = acc0[r] + bv;
        output[(size_t)b0 * NN + n] = v0 > 0.f ? v0 : 0.f;
        int b1 = (bTile0 + 1) * 16 + r + 8 * half;
        float v1 = acc1[r] + bv;
        output[(size_t)b1 * NN + n] = v1 > 0.f ? v1 : 0.f;
    }
}

// ---------------------------------------------------------------------------
// Backward: out_der[b,n,q] = (output[b,n] > 0) * sum_d K[d,n] * P[b,d,q]
// One WG = 4 batches. P slabs staged in LDS as bf16, transposed to [q][d].
// PSTRIDE = 520 elems -> row stride 260 dwords == 4 (mod 64): the 16 q-rows
// of a fragment map to 16 distinct 4-bank groups => ds_read_b128 with zero
// bank conflicts (2 phases = minimum for 512B/instr). Rows are 16B-aligned,
// so fragments load as two aligned v8bf halves.
// ---------------------------------------------------------------------------
#define PSTRIDE 520
#define PSLAB   (QQ * PSTRIDE)   // ushorts per batch slab (16640)

__global__ void __launch_bounds__(256)
backward_gemm(const float* __restrict__ prev_der,
              const unsigned short* __restrict__ Asw,
              const float* __restrict__ output,
              float* __restrict__ outder) {
    extern __shared__ __align__(32) unsigned char smem[];
    unsigned short* pt   = (unsigned short*)smem;          // 4 * PSLAB ushorts
    float* maskv         = (float*)(smem + 4 * PSLAB * 2); // 4 * NN floats

    int tid   = threadIdx.x;
    int bbase = blockIdx.x * 4;

    // Stage prev_der -> LDS (bf16, transposed [q][d]) as 2x2 patches:
    // two coalesced float2 reads + two packed dword LDS writes per patch.
#pragma unroll
    for (int bi = 0; bi < 4; ++bi) {
        const float* src = prev_der + (size_t)(bbase + bi) * (DD * QQ);
        unsigned short* dst = pt + bi * PSLAB;
        for (int p = tid; p < (DD / 2) * (QQ / 2); p += 256) {
            int d = (p >> 4) * 2;            // 0,2,..,510
            int q = (p & 15) * 2;            // 0,2,..,30
            float2 r0 = *(const float2*)(src + d * QQ + q);        // row d
            float2 r1 = *(const float2*)(src + (d + 1) * QQ + q);  // row d+1
            unsigned int w0 = (unsigned int)f2bf(r0.x) | ((unsigned int)f2bf(r1.x) << 16);
            unsigned int w1 = (unsigned int)f2bf(r0.y) | ((unsigned int)f2bf(r1.y) << 16);
            *(unsigned int*)(dst + q * PSTRIDE + d)       = w0;    // row q: (d,d+1)
            *(unsigned int*)(dst + (q + 1) * PSTRIDE + d) = w1;    // row q+1
        }
    }
    // Stage relu-mask source (forward activations).
    for (int j = tid; j < 4 * NN; j += 256) {
        int bi = j >> 9, n = j & (NN - 1);
        maskv[j] = output[(size_t)(bbase + bi) * NN + n];
    }
    __syncthreads();

    int w    = tid >> 5;
    int lane = tid & 31;
    int half = lane >> 4;
    int nl   = lane & 15;

    for (int ntI = 0; ntI < 4; ++ntI) {
        int nt = w + ntI * 8;   // each wave owns 4 of the 32 n-tiles

        v8f acc[4][2];
#pragma unroll
        for (int bi = 0; bi < 4; ++bi)
#pragma unroll
            for (int qt = 0; qt < 2; ++qt)
                acc[bi][qt] = (v8f){0.f, 0.f, 0.f, 0.f, 0.f, 0.f, 0.f, 0.f};

        for (int kc = 0; kc < 16; ++kc) {
            // A fragment (K^T tile) from L2-hot swizzled buffer, reused 8x
            v16bf afrag = *(const v16bf*)(Asw + (((nt * 16 + kc) * 32 + lane) << 4));
#pragma unroll
            for (int bi = 0; bi < 4; ++bi) {
                const unsigned short* pbase = pt + bi * PSLAB + kc * 32 + half * 16;
#pragma unroll
                for (int qt = 0; qt < 2; ++qt) {
                    const unsigned short* pr = pbase + (qt * 16 + nl) * PSTRIDE;
                    v8bf blo = *(const v8bf*)(pr);
                    v8bf bhi = *(const v8bf*)(pr + 8);
                    v16bf bfrag = __builtin_shufflevector(
                        blo, bhi, 0, 1, 2, 3, 4, 5, 6, 7, 8, 9, 10, 11, 12, 13, 14, 15);
                    acc[bi][qt] = __builtin_amdgcn_wmma_f32_16x16x32_bf16(
                        false, afrag, false, bfrag, (short)0, acc[bi][qt], false, false);
                }
            }
        }

        // Epilogue: mask by g'(a) = (output > 0), store f32.
#pragma unroll
        for (int bi = 0; bi < 4; ++bi) {
#pragma unroll
            for (int qt = 0; qt < 2; ++qt) {
#pragma unroll
                for (int r = 0; r < 8; ++r) {
                    int n = nt * 16 + r + 8 * half;
                    float g = maskv[bi * NN + n];
                    float v = (g > 0.f) ? acc[bi][qt][r] : 0.f;
                    outder[((size_t)(bbase + bi) * NN + n) * QQ + qt * 16 + nl] = v;
                }
            }
        }
    }
}

// ---------------------------------------------------------------------------
extern "C" void kernel_launch(void* const* d_in, const int* in_sizes, int n_in,
                              void* d_out, int out_size, void* d_ws, size_t ws_size,
                              hipStream_t stream) {
    const float* inputs   = (const float*)d_in[0];
    const float* prev_der = (const float*)d_in[1];
    const float* kern     = (const float*)d_in[2];
    const float* bias     = (const float*)d_in[3];

    float* output = (float*)d_out;                       // [B, N]
    float* outder = output + (size_t)BB * NN;            // [B, N, Q]

    // Workspace: Asw (512KB) | Bsw (512KB) | Isw (2MB) -- all bf16
    unsigned short* Asw = (unsigned short*)d_ws;
    unsigned short* Bsw = Asw + 32 * 16 * 32 * 16;       // 262144
    unsigned short* Isw = Bsw + 32 * 16 * 32 * 16;       // 262144

    swizzle_weights<<<1024, 256, 0, stream>>>(kern, Asw, Bsw);
    swizzle_inputs<<<4096, 256, 0, stream>>>(inputs, Isw);
    forward_gemm<<<dim3(BB / 64, NN / 64), 256, 0, stream>>>(Isw, Bsw, bias, output);

    size_t ldsBytes = (size_t)4 * PSLAB * 2 + (size_t)4 * NN * 4; // 141312 B
    backward_gemm<<<BB / 4, 256, ldsBytes, stream>>>(prev_der, Asw, output, outder);
}
